// GAT_AML_32246614458738
// MI455X (gfx1250) — matrix operand order repaired and checked by reading, not compiled, not run on hardware
//
#include <hip/hip_runtime.h>
#include <math.h>

// ---------------------------------------------------------------------------
// CDNA5 (gfx1250) GAT pipeline: bf16 WMMA GEMMs (LDS-staged B) + atomic
// segment ops for the edge-softmax/aggregation phase.
// ---------------------------------------------------------------------------

typedef __attribute__((ext_vector_type(16))) __bf16 v16bf;
typedef __attribute__((ext_vector_type(8)))  float  v8f;

union BF16x16 { v16bf v; unsigned short u[16]; uint4 q[2]; };

__device__ __forceinline__ unsigned short f2bf(float f) {
  unsigned u = __float_as_uint(f);
  u += 0x7FFFu + ((u >> 16) & 1u);          // round-to-nearest-even
  return (unsigned short)(u >> 16);
}

// monotonic float <-> uint encoding for atomicMax-based segment max
__device__ __forceinline__ unsigned fenc(float f) {
  unsigned u = __float_as_uint(f);
  return (u & 0x80000000u) ? ~u : (u | 0x80000000u);
}
__device__ __forceinline__ float fdec(unsigned u) {
  unsigned b = (u & 0x80000000u) ? (u & 0x7FFFFFFFu) : ~u;
  return __uint_as_float(b);
}
__device__ __forceinline__ float lrelu(float x) { return x > 0.f ? x : 0.2f * x; }

#define ENC_NEG_INF 0x007FFFFFu   // fenc(-inf)

// ---------------------------------------------------------------------------
// C[M,Nc] = A[M,K] * B[K,Nc]; fp32 in memory, bf16 WMMA, f32 accumulate.
// K, Nc compile-time.  Block = 256 threads = 8 waves.
//   - B converted to bf16 and staged in LDS *transposed* ([n][k], stride K+8
//     -> conflict-free ds_load_b128 fragment reads).
//   - Each wave owns one 16-row stripe: A fragments preloaded once
//     (float4 global loads), reused across all Nc/16 column tiles.
// A frag layout (ISA 7.12.2): elem e<8 -> K=ka+e, e>=8 -> K=ka+16+(e-8),
//   ka = 32*ks + 8*half.  B frag: same K mapping as A, col=l.
// C/D: VGPR r -> row = 16*tm + 8*half + r, col = l.
// ---------------------------------------------------------------------------
template <int K, int Nc>
__global__ __launch_bounds__(256)
void gemm_bf16_wmma(const float* __restrict__ A, const float* __restrict__ B,
                    float* __restrict__ C, int M) {
  constexpr int KS  = K / 32;     // WMMA k-steps
  constexpr int TN  = Nc / 16;    // column tiles
  constexpr int LDK = K + 8;      // padded LDS stride (ushorts)
  __shared__ unsigned short ldsB[Nc * LDK];

  // ---- cooperative fill: B row-major fp32 [K][Nc] -> LDS bf16 [n][k] ----
  for (int i = threadIdx.x; i < K * Nc; i += 256) {
    int k = i / Nc, n = i - k * Nc;
    ldsB[n * LDK + k] = f2bf(B[i]);
  }
  __syncthreads();

  int tm     = blockIdx.x * 8 + (threadIdx.x >> 5);
  int tilesM = (M + 15) >> 4;
  if (tm >= tilesM) return;                 // wave-uniform; EXEC all-1 below

  int lane = threadIdx.x & 31;
  int half = lane >> 4;
  int l    = lane & 15;

  int rowA = tm * 16 + l; if (rowA >= M) rowA = M - 1;   // clamped load
  const float* Ar = A + (size_t)rowA * K;

  // ---- preload all A fragments for this 16-row stripe ----
  BF16x16 afrag[KS];
#pragma unroll
  for (int ks = 0; ks < KS; ++ks) {
    int ka = ks * 32 + half * 8;
    float4 f0 = *(const float4*)(Ar + ka);
    float4 f1 = *(const float4*)(Ar + ka + 4);
    float4 f2 = *(const float4*)(Ar + ka + 16);
    float4 f3 = *(const float4*)(Ar + ka + 20);
    afrag[ks].u[0]  = f2bf(f0.x); afrag[ks].u[1]  = f2bf(f0.y);
    afrag[ks].u[2]  = f2bf(f0.z); afrag[ks].u[3]  = f2bf(f0.w);
    afrag[ks].u[4]  = f2bf(f1.x); afrag[ks].u[5]  = f2bf(f1.y);
    afrag[ks].u[6]  = f2bf(f1.z); afrag[ks].u[7]  = f2bf(f1.w);
    afrag[ks].u[8]  = f2bf(f2.x); afrag[ks].u[9]  = f2bf(f2.y);
    afrag[ks].u[10] = f2bf(f2.z); afrag[ks].u[11] = f2bf(f2.w);
    afrag[ks].u[12] = f2bf(f3.x); afrag[ks].u[13] = f2bf(f3.y);
    afrag[ks].u[14] = f2bf(f3.z); afrag[ks].u[15] = f2bf(f3.w);
  }

  const bool fullM = (tm * 16 + 16) <= M;
  const int  rbase = tm * 16 + half * 8;

#pragma unroll
  for (int tn = 0; tn < TN; ++tn) {
    v8f acc = {};
#pragma unroll
    for (int ks = 0; ks < KS; ++ks) {
      BF16x16 b;
      const uint4* bp =
          (const uint4*)&ldsB[(tn * 16 + l) * LDK + ks * 32 + half * 8];
      b.q[0] = bp[0];
      b.q[1] = bp[2];
      acc = __builtin_amdgcn_wmma_f32_16x16x32_bf16(
          /*neg_a=*/false, afrag[ks].v, /*neg_b=*/false, b.v,
          /*c_mod=*/(short)0, acc, /*reuse_a=*/false, /*reuse_b=*/false);
    }
    int col = tn * 16 + l;
    if (fullM) {
#pragma unroll
      for (int r = 0; r < 8; ++r)
        C[(rbase + r) * Nc + col] = acc[r];
    } else {
#pragma unroll
      for (int r = 0; r < 8; ++r)
        if (rbase + r < M) C[(rbase + r) * Nc + col] = acc[r];
    }
  }
}

// ---------------------------------------------------------------------------
// alpha[n*H+h] = dot(Hlin[n, h*64 : h*64+64], avec[h, :])   (wave per (n,h))
// ---------------------------------------------------------------------------
__global__ void attn_alpha(const float* __restrict__ Hlin,
                           const float* __restrict__ avec,
                           float* __restrict__ out, int Nn, int H) {
  int wid  = (blockIdx.x * blockDim.x + threadIdx.x) >> 5;
  int lane = threadIdx.x & 31;
  if (wid >= Nn * H) return;
  int n = wid / H, h = wid % H;
  const float* hp = Hlin + ((size_t)n * H + h) * 64;
  const float* ap = avec + h * 64;
  float v = hp[lane] * ap[lane] + hp[lane + 32] * ap[lane + 32];
#pragma unroll
  for (int off = 16; off > 0; off >>= 1) v += __shfl_down(v, off, 32);
  if (lane == 0) out[wid] = v;
}

__global__ void fill_u32(unsigned* __restrict__ p, unsigned v, int n) {
  int i = blockIdx.x * blockDim.x + threadIdx.x;
  if (i < n) p[i] = v;
}

// ---------------------------------------------------------------------------
// Edge passes (self-loops appended implicitly: edge i>=E has src=dst=i-E).
// ---------------------------------------------------------------------------
__global__ void edge_max(const int* __restrict__ ei, int E, int Nn,
                         const float* __restrict__ as, const float* __restrict__ ad,
                         unsigned* __restrict__ m, int H) {
  int idx = blockIdx.x * blockDim.x + threadIdx.x;
  int nE = E + Nn;
  if (idx >= nE * H) return;
  int h = idx % H, i = idx / H;
  int s = (i < E) ? ei[i]     : (i - E);
  int d = (i < E) ? ei[E + i] : (i - E);
  float e = lrelu(as[s * H + h] + ad[d * H + h]);
  atomicMax(&m[d * H + h], fenc(e));
}

__global__ void edge_sum(const int* __restrict__ ei, int E, int Nn,
                         const float* __restrict__ as, const float* __restrict__ ad,
                         const unsigned* __restrict__ m, float* __restrict__ denom, int H) {
  int idx = blockIdx.x * blockDim.x + threadIdx.x;
  int nE = E + Nn;
  if (idx >= nE * H) return;
  int h = idx % H, i = idx / H;
  int s = (i < E) ? ei[i]     : (i - E);
  int d = (i < E) ? ei[E + i] : (i - E);
  float e  = lrelu(as[s * H + h] + ad[d * H + h]);
  float ex = expf(e - fdec(m[d * H + h]));
  atomicAdd(&denom[d * H + h], ex);
}

// one thread per (edge, head, 4-channel chunk); float4 gather + 4 fp32 atomics
__global__ void edge_agg(const int* __restrict__ ei, int E, int Nn,
                         const float* __restrict__ as, const float* __restrict__ ad,
                         const unsigned* __restrict__ m, const float* __restrict__ denom,
                         const float* __restrict__ Hlin, float* __restrict__ agg, int H) {
  long long idx = (long long)blockIdx.x * blockDim.x + threadIdx.x;
  int c4 = (int)(idx & 15);
  long long eh = idx >> 4;
  long long tot = (long long)(E + Nn) * H;
  if (eh >= tot) return;
  int h = (int)(eh % H);
  long long i = eh / H;
  int s = (i < E) ? ei[i]     : (int)(i - E);
  int d = (i < E) ? ei[E + i] : (int)(i - E);
  float e = lrelu(as[s * H + h] + ad[d * H + h]);
  float w = expf(e - fdec(m[d * H + h])) / (denom[d * H + h] + 1e-16f);
  const float4 hv = *(const float4*)(Hlin + ((size_t)s * H + h) * 64 + c4 * 4);
  float* ap = agg + ((size_t)d * H + h) * 64 + c4 * 4;
  atomicAdd(ap + 0, hv.x * w);
  atomicAdd(ap + 1, hv.y * w);
  atomicAdd(ap + 2, hv.z * w);
  atomicAdd(ap + 3, hv.w * w);
}

// out[n,c] = ELU( mean_h agg[n,h,c] + bias[c] )
__global__ void node_finalize(const float* __restrict__ agg, const float* __restrict__ bias,
                              float* __restrict__ out, int Nn, int H) {
  int idx = blockIdx.x * blockDim.x + threadIdx.x;
  if (idx >= Nn * 64) return;
  int n = idx >> 6, c = idx & 63;
  float ssum = 0.f;
  for (int h = 0; h < H; ++h) ssum += agg[((size_t)n * H + h) * 64 + c];
  ssum = ssum / (float)H + bias[c];
  out[idx] = ssum > 0.f ? ssum : expm1f(ssum);
}

// logits = h2 @ Wc + bc (Wc is [64,2] row-major), then log_softmax over 2
__global__ void classifier(const float* __restrict__ h2, const float* __restrict__ Wc,
                           const float* __restrict__ bc, float* __restrict__ out, int Nn) {
  int wid  = (blockIdx.x * blockDim.x + threadIdx.x) >> 5;
  int lane = threadIdx.x & 31;
  if (wid >= Nn) return;
  const float* hp = h2 + (size_t)wid * 64;
  float f0 = hp[lane], f1 = hp[lane + 32];
  float l0 = f0 * Wc[lane * 2]     + f1 * Wc[(lane + 32) * 2];
  float l1 = f0 * Wc[lane * 2 + 1] + f1 * Wc[(lane + 32) * 2 + 1];
#pragma unroll
  for (int off = 16; off > 0; off >>= 1) {
    l0 += __shfl_down(l0, off, 32);
    l1 += __shfl_down(l1, off, 32);
  }
  if (lane == 0) {
    l0 += bc[0]; l1 += bc[1];
    float mx  = fmaxf(l0, l1);
    float lse = mx + logf(expf(l0 - mx) + expf(l1 - mx));
    out[wid * 2 + 0] = l0 - lse;
    out[wid * 2 + 1] = l1 - lse;
  }
}

// ---------------------------------------------------------------------------
extern "C" void kernel_launch(void* const* d_in, const int* in_sizes, int n_in,
                              void* d_out, int out_size, void* d_ws, size_t ws_size,
                              hipStream_t stream) {
  const float* x      = (const float*)d_in[0];
  const int*   ei     = (const int*)  d_in[1];
  const float* W1     = (const float*)d_in[2];
  const float* a_src1 = (const float*)d_in[3];
  const float* a_dst1 = (const float*)d_in[4];
  const float* b1     = (const float*)d_in[5];
  const float* W2     = (const float*)d_in[6];
  const float* a_src2 = (const float*)d_in[7];
  const float* a_dst2 = (const float*)d_in[8];
  const float* b2     = (const float*)d_in[9];
  const float* Wc     = (const float*)d_in[10];
  const float* bc     = (const float*)d_in[11];
  float* out = (float*)d_out;

  const int D = 128;
  const int Nn = in_sizes[0] / D;     // 50000
  const int E  = in_sizes[1] / 2;     // 800000
  const int nE = E + Nn;              // with self-loops
  const int tilesM = (Nn + 15) / 16;

  // ---- workspace layout (floats) ----
  float*    ws   = (float*)d_ws;
  float*    Hlin = ws;                          // N*128 (L1) / N*64 (L2)
  float*    agg  = ws + (size_t)Nn * 128;       // N*128 (L1) / N*64 (L2)
  float*    hbuf = ws + (size_t)Nn * 256;       // N*64 : h1, later h2
  float*    sc   = ws + (size_t)Nn * 320;
  float*    as1  = sc;                          // N*2
  float*    ad1  = sc + (size_t)Nn * 2;         // N*2
  unsigned* m1   = (unsigned*)(sc + (size_t)Nn * 4);  // N*2
  float*    dn1  = sc + (size_t)Nn * 6;         // N*2
  float*    as2  = sc + (size_t)Nn * 8;         // N
  float*    ad2  = sc + (size_t)Nn * 9;         // N
  unsigned* m2   = (unsigned*)(sc + (size_t)Nn * 10); // N
  float*    dn2  = sc + (size_t)Nn * 11;        // N

  const int BLK = 256;
  auto cdiv = [](long long a, long long b) { return (unsigned)((a + b - 1) / b); };

  // ================= Layer 1 (H = 2) =================
  {
    const int H = 2;
    gemm_bf16_wmma<128, 128><<<cdiv(tilesM, 8), BLK, 0, stream>>>(x, W1, Hlin, Nn);

    attn_alpha<<<cdiv((long long)Nn * H, BLK / 32), BLK, 0, stream>>>(Hlin, a_src1, as1, Nn, H);
    attn_alpha<<<cdiv((long long)Nn * H, BLK / 32), BLK, 0, stream>>>(Hlin, a_dst1, ad1, Nn, H);

    fill_u32<<<cdiv((long long)Nn * H, BLK), BLK, 0, stream>>>(m1, ENC_NEG_INF, Nn * H);
    fill_u32<<<cdiv((long long)Nn * H, BLK), BLK, 0, stream>>>((unsigned*)dn1, 0u, Nn * H);
    fill_u32<<<cdiv((long long)Nn * 128, BLK), BLK, 0, stream>>>((unsigned*)agg, 0u, Nn * 128);

    edge_max<<<cdiv((long long)nE * H, BLK), BLK, 0, stream>>>(ei, E, Nn, as1, ad1, m1, H);
    edge_sum<<<cdiv((long long)nE * H, BLK), BLK, 0, stream>>>(ei, E, Nn, as1, ad1, m1, dn1, H);
    edge_agg<<<cdiv((long long)nE * H * 16, BLK), BLK, 0, stream>>>(ei, E, Nn, as1, ad1, m1, dn1,
                                                                    Hlin, agg, H);
    node_finalize<<<cdiv((long long)Nn * 64, BLK), BLK, 0, stream>>>(agg, b1, hbuf, Nn, H);
  }

  // ================= Layer 2 (H = 1) =================
  {
    const int H = 1;
    gemm_bf16_wmma<64, 64><<<cdiv(tilesM, 8), BLK, 0, stream>>>(hbuf, W2, Hlin, Nn);

    attn_alpha<<<cdiv((long long)Nn, BLK / 32), BLK, 0, stream>>>(Hlin, a_src2, as2, Nn, H);
    attn_alpha<<<cdiv((long long)Nn, BLK / 32), BLK, 0, stream>>>(Hlin, a_dst2, ad2, Nn, H);

    fill_u32<<<cdiv((long long)Nn, BLK), BLK, 0, stream>>>(m2, ENC_NEG_INF, Nn);
    fill_u32<<<cdiv((long long)Nn, BLK), BLK, 0, stream>>>((unsigned*)dn2, 0u, Nn);
    fill_u32<<<cdiv((long long)Nn * 64, BLK), BLK, 0, stream>>>((unsigned*)agg, 0u, Nn * 64);

    edge_max<<<cdiv((long long)nE, BLK), BLK, 0, stream>>>(ei, E, Nn, as2, ad2, m2, H);
    edge_sum<<<cdiv((long long)nE, BLK), BLK, 0, stream>>>(ei, E, Nn, as2, ad2, m2, dn2, H);
    edge_agg<<<cdiv((long long)nE * 16, BLK), BLK, 0, stream>>>(ei, E, Nn, as2, ad2, m2, dn2,
                                                                Hlin, agg, H);
    node_finalize<<<cdiv((long long)Nn * 64, BLK), BLK, 0, stream>>>(agg, b2, hbuf, Nn, H);
  }

  // ================= Classifier + log_softmax =================
  classifier<<<cdiv((long long)Nn, BLK / 32), BLK, 0, stream>>>(hbuf, Wc, bc, out, Nn);
}
